// PatchTSTEncoderLayerWithRope_29987461660825
// MI455X (gfx1250) — compile-verified, matrix operand order checked by reading, not run on hardware
//
#include <hip/hip_runtime.h>
#include <hip/hip_bf16.h>

// ---------------------------------------------------------------------------
// PatchTST encoder layer for gfx1250 (MI455X).
// GEMMs + attention via v_wmma_f32_16x16x32_bf16 (wave32 WMMA).
// Tile staging via GLOBAL_LOAD_ASYNC_TO_LDS_B128 + ASYNCcnt double-buffering.
// ---------------------------------------------------------------------------

typedef __attribute__((ext_vector_type(16))) __bf16 v16bf;
typedef __attribute__((ext_vector_type(8)))  float  v8f;

#define D_      512
#define S_      512
#define H_      8
#define HD_     64
#define FFN_    2048
#define ROWS_   32768   // B*C*S = 64*512

__device__ __forceinline__ __bf16 f2bf(float f) {
    unsigned u = __builtin_bit_cast(unsigned, f);
    unsigned r = u + 0x7FFFu + ((u >> 16) & 1u);      // round-to-nearest-even
    unsigned short h = (unsigned short)(r >> 16);
    return __builtin_bit_cast(__bf16, h);
}

// K-dimension element pattern for 16-bit A/B fragments (ISA 7.12.2):
// lane group g = lane/16; element e -> K = (e>=8?16:0) + g*8 + (e&7)
__device__ __forceinline__ int kpat(int e, int g) {
    return ((e >> 3) << 4) + (g << 3) + (e & 7);
}

// LDS byte offset of a generic pointer to a __shared__ object
__device__ __forceinline__ unsigned lds_off(const void* p) {
    return (unsigned)(unsigned long long)
        (__attribute__((address_space(3))) const void*)p;
}

// Async DMA: 16 bytes / lane, global -> LDS, tracked by ASYNCcnt
__device__ __forceinline__ void async_ld_b128(unsigned lds_byte_off, const void* gptr) {
    asm volatile("global_load_async_to_lds_b128 %0, %1, off"
                 :: "v"(lds_byte_off), "v"(gptr) : "memory");
}

// ---------------------------------------------------------------------------
// Weight convert: W (K x N) f32  ->  WT (N x K) bf16
// ---------------------------------------------------------------------------
__global__ void wtconv_kernel(const float* __restrict__ W, __bf16* __restrict__ WT,
                              int K, int N) {
    int idx = blockIdx.x * 256 + threadIdx.x;
    if (idx >= K * N) return;
    int k = idx / N, n = idx % N;
    WT[(size_t)n * K + k] = f2bf(W[idx]);
}

// ---------------------------------------------------------------------------
// LayerNorm over D=512, writes bf16 operand matrix.
// ---------------------------------------------------------------------------
__global__ __launch_bounds__(256)
void ln_kernel(const float* __restrict__ x, const float* __restrict__ g,
               const float* __restrict__ b, __bf16* __restrict__ out) {
    __shared__ float s1[256], s2[256];
    const int row = blockIdx.x, tid = threadIdx.x;
    const float* p = x + (size_t)row * D_;
    float v0 = p[tid], v1 = p[tid + 256];
    s1[tid] = v0 + v1;
    s2[tid] = v0 * v0 + v1 * v1;
    __syncthreads();
    for (int off = 128; off > 0; off >>= 1) {
        if (tid < off) { s1[tid] += s1[tid + off]; s2[tid] += s2[tid + off]; }
        __syncthreads();
    }
    float mean = s1[0] * (1.0f / D_);
    float var  = s2[0] * (1.0f / D_) - mean * mean;
    float rstd = rsqrtf(var + 1e-5f);
    out[(size_t)row * D_ + tid]       = f2bf((v0 - mean) * rstd * g[tid] + b[tid]);
    out[(size_t)row * D_ + tid + 256] = f2bf((v1 - mean) * rstd * g[tid + 256] + b[tid + 256]);
}

// ---------------------------------------------------------------------------
// Generic WMMA GEMM:  C[M,N] = A[M,K](bf16) @ BT[N,K]^T(bf16) + bias
// Block tile 128x64, 8 waves, each wave 16x64 (4 accumulators), K-step 32.
// Double-buffered LDS, async global->LDS staging (3 async ops / wave / stage).
// mode 0: f32 out (+ residual)        mode 1: Q  (scale + rope -> bf16 perhead)
// mode 2: K  (rope -> bf16 perhead)   mode 3: V  (bf16 perhead)
// mode 4: GELU -> bf16 row-major
// ---------------------------------------------------------------------------
__global__ __launch_bounds__(256)
void gemm_wmma_kernel(const __bf16* __restrict__ A, const __bf16* __restrict__ BT,
                      const float* __restrict__ bias, const float* __restrict__ res,
                      float* __restrict__ Cf, __bf16* __restrict__ Cb,
                      int M, int N, int K, int mode) {
    __shared__ __align__(16) __bf16 As[2][128 * 32];
    __shared__ __align__(16) __bf16 Bs[2][64 * 32];
    const int tid = threadIdx.x;
    const int wave = tid >> 5, lane = tid & 31;
    const int lane16 = lane & 15, g = lane >> 4;
    const int m0 = blockIdx.x * 128, n0 = blockIdx.y * 64;

    // per-thread staging slots
    const int ar0 = tid >> 2,          as0 = (tid & 3) * 8;          // A slot 0
    const int ar1 = (tid + 256) >> 2,  as1 = ((tid + 256) & 3) * 8;  // A slot 1
    const int br  = tid >> 2,          bs  = (tid & 3) * 8;          // B slot

    v8f acc[4];
#pragma unroll
    for (int t = 0; t < 4; ++t)
#pragma unroll
        for (int r = 0; r < 8; ++r) acc[t][r] = 0.0f;

    const int nk = K >> 5;

    // prologue: stage 0
    async_ld_b128(lds_off(&As[0][ar0 * 32 + as0]), A + (size_t)(m0 + ar0) * K + as0);
    async_ld_b128(lds_off(&As[0][ar1 * 32 + as1]), A + (size_t)(m0 + ar1) * K + as1);
    async_ld_b128(lds_off(&Bs[0][br * 32 + bs]),  BT + (size_t)(n0 + br) * K + bs);

    for (int kt = 0; kt < nk; ++kt) {
        const int cur = kt & 1;
        if (kt + 1 < nk) {
            const int nxt = cur ^ 1;
            const int k1 = (kt + 1) << 5;
            async_ld_b128(lds_off(&As[nxt][ar0 * 32 + as0]),
                          A + (size_t)(m0 + ar0) * K + k1 + as0);
            async_ld_b128(lds_off(&As[nxt][ar1 * 32 + as1]),
                          A + (size_t)(m0 + ar1) * K + k1 + as1);
            async_ld_b128(lds_off(&Bs[nxt][br * 32 + bs]),
                          BT + (size_t)(n0 + br) * K + k1 + bs);
            asm volatile("s_wait_asynccnt 3" ::: "memory");  // stage kt retired
        } else {
            asm volatile("s_wait_asynccnt 0" ::: "memory");
        }
        __syncthreads();

        v16bf a;
#pragma unroll
        for (int e = 0; e < 16; ++e)
            a[e] = As[cur][(wave * 16 + lane16) * 32 + kpat(e, g)];
#pragma unroll
        for (int t = 0; t < 4; ++t) {
            v16bf b;
#pragma unroll
            for (int e = 0; e < 16; ++e)
                b[e] = Bs[cur][(t * 16 + lane16) * 32 + kpat(e, g)];
            acc[t] = __builtin_amdgcn_wmma_f32_16x16x32_bf16(
                false, a, false, b, (short)0, acc[t], false, false);
        }
        __syncthreads();   // buffer `cur` free for stage kt+2
    }

    float bv[4];
#pragma unroll
    for (int t = 0; t < 4; ++t) bv[t] = bias ? bias[n0 + t * 16 + lane16] : 0.0f;
    const int rowBase = m0 + wave * 16 + g * 8;

    if (mode == 0) {
#pragma unroll
        for (int t = 0; t < 4; ++t)
#pragma unroll
            for (int r = 0; r < 8; ++r) {
                int row = rowBase + r;
                size_t idx = (size_t)row * N + n0 + t * 16 + lane16;
                float v = acc[t][r] + bv[t];
                if (res) v += res[idx];
                Cf[idx] = v;
            }
    } else if (mode == 4) {
#pragma unroll
        for (int t = 0; t < 4; ++t)
#pragma unroll
            for (int r = 0; r < 8; ++r) {
                int row = rowBase + r;
                size_t idx = (size_t)row * N + n0 + t * 16 + lane16;
                float v = acc[t][r] + bv[t];
                v = 0.5f * v * (1.0f + erff(v * 0.7071067811865476f));  // exact GELU
                Cb[idx] = f2bf(v);
            }
    } else {
        const float scale = (mode == 1) ? 0.125f : 1.0f;  // HD^-0.5 = 1/8
#pragma unroll
        for (int t = 0; t < 4; ++t)
#pragma unroll
            for (int r = 0; r < 8; ++r) acc[t][r] = (acc[t][r] + bv[t]) * scale;
        if (mode != 3) {
            // partial RoPE: d<16 rotates, d in [16,32) has timescale inf -> identity
            const float LOG_WL = 9.210340371976184f;  // ln(10000)
#pragma unroll
            for (int t = 0; t < 2; ++t) {
                int d1 = t * 16 + lane16;                 // 0..31
                float inv_ts = (d1 < 16)
                    ? __expf(-LOG_WL * (float)(2 * d1) * (1.0f / 64.0f)) : 0.0f;
#pragma unroll
                for (int r = 0; r < 8; ++r) {
                    int srow = (rowBase + r) & (S_ - 1);
                    float ang = (float)srow * inv_ts;
                    float sn = __sinf(ang), cs = __cosf(ang);
                    float x1 = acc[t][r], x2 = acc[t + 2][r];
                    acc[t][r]     = x1 * cs - x2 * sn;
                    acc[t + 2][r] = x2 * cs + x1 * sn;
                }
            }
        }
        // per-head bf16 store: [(bc*H + h)*S + s]*HD + d
        const int h = n0 >> 6;
#pragma unroll
        for (int t = 0; t < 4; ++t)
#pragma unroll
            for (int r = 0; r < 8; ++r) {
                int row = rowBase + r;
                int bc = row >> 9, s = row & (S_ - 1);
                size_t idx = (((size_t)(bc * H_ + h)) * S_ + s) * HD_ + t * 16 + lane16;
                Cb[idx] = f2bf(acc[t][r]);
            }
    }
}

// ---------------------------------------------------------------------------
// Flash-attention: block = (head bh, 128 q-rows). 8 waves x 16 q-rows each.
// K/V streamed in double-buffered 64x64 LDS tiles via async DMA;
// QK^T and P@V via WMMA; online softmax with 16-lane shfl reductions.
// ---------------------------------------------------------------------------
__global__ __launch_bounds__(256)
void attn_kernel(const __bf16* __restrict__ Q, const __bf16* __restrict__ Kb,
                 const __bf16* __restrict__ Vb, __bf16* __restrict__ O) {
    __shared__ __align__(16) __bf16 Kt[2][64 * 64];
    __shared__ __align__(16) __bf16 Vt[2][64 * 64];
    __shared__ __align__(16) __bf16 Pt[8 * 16 * 64];   // per-wave P strip
    const int tid = threadIdx.x;
    const int wave = tid >> 5, lane = tid & 31;
    const int lane16 = lane & 15, g = lane >> 4;
    const int bh = blockIdx.y;
    const int q0 = blockIdx.x * 128;

    const __bf16* qh = Q  + (size_t)bh * S_ * HD_;
    const __bf16* kh = Kb + (size_t)bh * S_ * HD_;
    const __bf16* vh = Vb + (size_t)bh * S_ * HD_;

    // Q fragments (16 rows x 64, as two 16x32 A-fragments), kept in registers
    v16bf qa[2];
#pragma unroll
    for (int f = 0; f < 2; ++f)
#pragma unroll
        for (int e = 0; e < 16; ++e)
            qa[f][e] = qh[(size_t)(q0 + wave * 16 + lane16) * HD_ + f * 32 + kpat(e, g)];

    v8f o[4];
    float m_run[8], l_run[8];
#pragma unroll
    for (int t = 0; t < 4; ++t)
#pragma unroll
        for (int r = 0; r < 8; ++r) o[t][r] = 0.0f;
#pragma unroll
    for (int r = 0; r < 8; ++r) { m_run[r] = -1e30f; l_run[r] = 0.0f; }

    // prologue: stage tile 0 (8 bf16 per thread per tensor)
    async_ld_b128(lds_off(&Kt[0][tid * 8]), kh + (size_t)tid * 8);
    async_ld_b128(lds_off(&Vt[0][tid * 8]), vh + (size_t)tid * 8);

    for (int kt = 0; kt < 8; ++kt) {
        const int cur = kt & 1;
        if (kt + 1 < 8) {
            const int nxt = cur ^ 1;
            const size_t goff = (size_t)(kt + 1) * 64 * HD_ + tid * 8;
            async_ld_b128(lds_off(&Kt[nxt][tid * 8]), kh + goff);
            async_ld_b128(lds_off(&Vt[nxt][tid * 8]), vh + goff);
            asm volatile("s_wait_asynccnt 2" ::: "memory");  // tile kt retired
        } else {
            asm volatile("s_wait_asynccnt 0" ::: "memory");
        }
        __syncthreads();

        // scores = Q @ K^T : B^T == K rows (fast fragment form)
        v8f sc[4];
#pragma unroll
        for (int t = 0; t < 4; ++t)
#pragma unroll
            for (int r = 0; r < 8; ++r) sc[t][r] = 0.0f;
#pragma unroll
        for (int f = 0; f < 2; ++f) {
#pragma unroll
            for (int t = 0; t < 4; ++t) {
                v16bf b;
#pragma unroll
                for (int e = 0; e < 16; ++e)
                    b[e] = Kt[cur][(t * 16 + lane16) * HD_ + f * 32 + kpat(e, g)];
                sc[t] = __builtin_amdgcn_wmma_f32_16x16x32_bf16(
                    false, qa[f], false, b, (short)0, sc[t], false, false);
            }
        }

        // online softmax (row = r + 8*g; 16 cols of each tile across 16 lanes)
        float fac[8];
#pragma unroll
        for (int r = 0; r < 8; ++r) {
            float mx = fmaxf(fmaxf(sc[0][r], sc[1][r]), fmaxf(sc[2][r], sc[3][r]));
#pragma unroll
            for (int mask = 1; mask <= 8; mask <<= 1)
                mx = fmaxf(mx, __shfl_xor(mx, mask, 32));
            float mnew = fmaxf(m_run[r], mx);
            fac[r] = __expf(m_run[r] - mnew);
            m_run[r] = mnew;
            float rs = 0.0f;
#pragma unroll
            for (int t = 0; t < 4; ++t) {
                float p = __expf(sc[t][r] - mnew);
                sc[t][r] = p;
                rs += p;
            }
#pragma unroll
            for (int mask = 1; mask <= 8; mask <<= 1)
                rs += __shfl_xor(rs, mask, 32);
            l_run[r] = l_run[r] * fac[r] + rs;
#pragma unroll
            for (int t = 0; t < 4; ++t) o[t][r] *= fac[r];
        }

        // P: C-layout -> LDS -> reload as A-fragments (per-wave region, no barrier)
#pragma unroll
        for (int t = 0; t < 4; ++t)
#pragma unroll
            for (int r = 0; r < 8; ++r)
                Pt[wave * 1024 + (r + g * 8) * 64 + t * 16 + lane16] = f2bf(sc[t][r]);

        // O += P @ V   (B in KxN form: Vt[k*64 + n])
#pragma unroll
        for (int f = 0; f < 2; ++f) {
            v16bf pa;
#pragma unroll
            for (int e = 0; e < 16; ++e)
                pa[e] = Pt[wave * 1024 + lane16 * 64 + f * 32 + kpat(e, g)];
#pragma unroll
            for (int t = 0; t < 4; ++t) {
                v16bf b;
#pragma unroll
                for (int e = 0; e < 16; ++e)
                    b[e] = Vt[cur][(f * 32 + kpat(e, g)) * 64 + t * 16 + lane16];
                o[t] = __builtin_amdgcn_wmma_f32_16x16x32_bf16(
                    false, pa, false, b, (short)0, o[t], false, false);
            }
        }
        __syncthreads();   // buffer `cur` free for tile kt+2
    }

    // normalize + store bf16 as (bc, s, D) row-major for the O-projection GEMM
    const int bc = bh >> 3, h = bh & 7;
#pragma unroll
    for (int r = 0; r < 8; ++r) {
        float inv = (l_run[r] > 0.0f) ? 1.0f / l_run[r] : 0.0f;
        int s = q0 + wave * 16 + r + g * 8;
#pragma unroll
        for (int t = 0; t < 4; ++t) {
            size_t idx = ((size_t)(bc * S_ + s)) * D_ + h * HD_ + t * 16 + lane16;
            O[idx] = f2bf(o[t][r] * inv);
        }
    }
}

// ---------------------------------------------------------------------------
// Host-side orchestration
// ---------------------------------------------------------------------------
extern "C" void kernel_launch(void* const* d_in, const int* in_sizes, int n_in,
                              void* d_out, int out_size, void* d_ws, size_t ws_size,
                              hipStream_t stream) {
    (void)in_sizes; (void)n_in; (void)out_size; (void)ws_size;
    const float* x     = (const float*)d_in[0];
    const float* Wq    = (const float*)d_in[1];
    const float* bq    = (const float*)d_in[2];
    const float* Wk    = (const float*)d_in[3];
    const float* bk    = (const float*)d_in[4];
    const float* Wv    = (const float*)d_in[5];
    const float* bv    = (const float*)d_in[6];
    const float* Wo    = (const float*)d_in[7];
    const float* bo    = (const float*)d_in[8];
    const float* ln1_g = (const float*)d_in[9];
    const float* ln1_b = (const float*)d_in[10];
    const float* ln3_g = (const float*)d_in[11];
    const float* ln3_b = (const float*)d_in[12];
    const float* W1    = (const float*)d_in[13];
    const float* b1    = (const float*)d_in[14];
    const float* W2    = (const float*)d_in[15];
    const float* b2    = (const float*)d_in[16];
    float* out = (float*)d_out;

    char* ws = (char*)d_ws;
    size_t off = 0;
    __bf16* WqT = (__bf16*)(ws + off); off += (size_t)D_ * D_ * 2;
    __bf16* WkT = (__bf16*)(ws + off); off += (size_t)D_ * D_ * 2;
    __bf16* WvT = (__bf16*)(ws + off); off += (size_t)D_ * D_ * 2;
    __bf16* WoT = (__bf16*)(ws + off); off += (size_t)D_ * D_ * 2;
    __bf16* W1T = (__bf16*)(ws + off); off += (size_t)D_ * FFN_ * 2;   // (FFN x D)
    __bf16* W2T = (__bf16*)(ws + off); off += (size_t)FFN_ * D_ * 2;   // (D x FFN)
    __bf16* hn  = (__bf16*)(ws + off); off += (size_t)ROWS_ * D_ * 2;  // LN out (reused)
    float*  h2  = (float*)(ws + off);  off += (size_t)ROWS_ * D_ * 4;  // residual stream
    char*   regA = ws + off;                                           // 128 MB region
    __bf16* qb   = (__bf16*)(regA);
    __bf16* kb   = (__bf16*)(regA + (size_t)ROWS_ * D_ * 2);
    __bf16* vb   = (__bf16*)(regA + (size_t)ROWS_ * D_ * 2 * 2);
    __bf16* ob   = (__bf16*)(regA + (size_t)ROWS_ * D_ * 2 * 3);
    __bf16* ffn1 = (__bf16*)(regA);   // reuses q/k/v/o after attention is consumed

    const dim3 blk(256);

    // 1) weight conversion/transposition (f32 -> bf16, N x K)
    wtconv_kernel<<<dim3((D_ * D_ + 255) / 256), blk, 0, stream>>>(Wq, WqT, D_, D_);
    wtconv_kernel<<<dim3((D_ * D_ + 255) / 256), blk, 0, stream>>>(Wk, WkT, D_, D_);
    wtconv_kernel<<<dim3((D_ * D_ + 255) / 256), blk, 0, stream>>>(Wv, WvT, D_, D_);
    wtconv_kernel<<<dim3((D_ * D_ + 255) / 256), blk, 0, stream>>>(Wo, WoT, D_, D_);
    wtconv_kernel<<<dim3((D_ * FFN_ + 255) / 256), blk, 0, stream>>>(W1, W1T, D_, FFN_);
    wtconv_kernel<<<dim3((FFN_ * D_ + 255) / 256), blk, 0, stream>>>(W2, W2T, FFN_, D_);

    // 2) LN1
    ln_kernel<<<dim3(ROWS_), blk, 0, stream>>>(x, ln1_g, ln1_b, hn);

    // 3) QKV projections with fused scale/RoPE/per-head bf16 repack
    const dim3 gD(ROWS_ / 128, D_ / 64);
    gemm_wmma_kernel<<<gD, blk, 0, stream>>>(hn, WqT, bq, nullptr, nullptr, qb,
                                             ROWS_, D_, D_, 1);
    gemm_wmma_kernel<<<gD, blk, 0, stream>>>(hn, WkT, bk, nullptr, nullptr, kb,
                                             ROWS_, D_, D_, 2);
    gemm_wmma_kernel<<<gD, blk, 0, stream>>>(hn, WvT, bv, nullptr, nullptr, vb,
                                             ROWS_, D_, D_, 3);

    // 4) attention (512 heads x 4 q-strips)
    attn_kernel<<<dim3(S_ / 128, 64 * H_), blk, 0, stream>>>(qb, kb, vb, ob);

    // 5) O-projection + residual -> h2 (f32)
    gemm_wmma_kernel<<<gD, blk, 0, stream>>>(ob, WoT, bo, x, h2, nullptr,
                                             ROWS_, D_, D_, 0);

    // 6) LN2
    ln_kernel<<<dim3(ROWS_), blk, 0, stream>>>(h2, ln3_g, ln3_b, hn);

    // 7) FFN1 with fused exact GELU -> bf16 (overwrites q/k/v/o region)
    gemm_wmma_kernel<<<dim3(ROWS_ / 128, FFN_ / 64), blk, 0, stream>>>(
        hn, W1T, b1, nullptr, nullptr, ffn1, ROWS_, FFN_, D_, 4);

    // 8) FFN2 + residual -> final f32 output
    gemm_wmma_kernel<<<gD, blk, 0, stream>>>(ffn1, W2T, b2, h2, out, nullptr,
                                             ROWS_, D_, FFN_, 0);
}
